// QuantumEquivariantBlock_14516989461113
// MI455X (gfx1250) — matrix operand order repaired and checked by reading, not compiled, not run on hardware
//
#include <hip/hip_runtime.h>
#include <hip/hip_bf16.h>
#include <math.h>
#include <stdint.h>

#define N_NODES 30000
#define N_EDGES 480000
#define HF 128

typedef __attribute__((ext_vector_type(16))) _Float16 v16h;
typedef __attribute__((ext_vector_type(8)))  float    v8f;

struct cplx { float re, im; };

__device__ __forceinline__ float silu_f(float x) { return x / (1.0f + __expf(-x)); }

union AFrag { v16h v; _Float16 h[16]; };

// A-matrix fragment (16x32 f16, MxK) from LDS, per CDNA5 ISA layout:
// lanes 0-15: row=lane, K = [k0..k0+7] in halves 0..7, [k0+16..k0+23] in halves 8..15
// lanes 16-31: row=lane-16, K = [k0+8..k0+15], [k0+24..k0+31]
__device__ __forceinline__ v16h load_a_frag(const _Float16* a, int stride, int k0, int lane) {
    int row  = lane & 15;
    int koff = k0 + ((lane >> 4) << 3);
    const _Float16* p = a + row * stride + koff;
    AFrag f;
#pragma unroll
    for (int i = 0; i < 8; ++i) { f.h[i] = p[i]; f.h[8 + i] = p[16 + i]; }
    return f.v;
}

// B fragments pre-swizzled: tile t, lane L, halves i=0..15 contiguous (32B/lane).
__device__ __forceinline__ v16h load_b_frag(const _Float16* base, int tile, int lane) {
    return *(const v16h*)(base + (size_t)tile * 512 + lane * 16);
}

// CDNA5 async global->LDS copy (16B per lane), tracked by ASYNCcnt.
__device__ __forceinline__ void async_ld_b128(uint32_t lds_addr, const void* gaddr) {
    asm volatile("global_load_async_to_lds_b128 %0, %1, off"
                 :: "v"(lds_addr), "v"(gaddr) : "memory");
}
__device__ __forceinline__ void wait_asynccnt0() {
    asm volatile("s_wait_asynccnt 0" ::: "memory");
}

// ---------------------------------------------------------------------------
// Weight swizzle: src [K x ld] f32 row-major -> per-lane B-fragment f16 layout.
// ---------------------------------------------------------------------------
__global__ void make_bfrag_kernel(const float* __restrict__ src, _Float16* __restrict__ dst,
                                  int Kt, int Nt, int ld) {
    size_t idx = (size_t)blockIdx.x * 256 + threadIdx.x;
    size_t total = (size_t)Kt * Nt * 512;
    if (idx >= total) return;
    int t = (int)(idx >> 9);
    int rem = (int)(idx & 511);
    int lane = rem >> 4, i = rem & 15;
    int kt = t / Nt, nt = t % Nt;
    int k = kt * 32 + ((lane >> 4) << 3) + (i & 7) + ((i >> 3) << 4);
    int n = nt * 16 + (lane & 15);
    dst[idx] = (_Float16)src[(size_t)k * ld + n];
}

__global__ void zero_kernel(float* p, int n) {
    int i = blockIdx.x * 256 + threadIdx.x;
    if (i < n) p[i] = 0.0f;
}

__global__ void to_half_kernel(const float* __restrict__ src, _Float16* __restrict__ dst, int n) {
    int i = blockIdx.x * 256 + threadIdx.x;
    if (i < n) dst[i] = (_Float16)src[i];
}

// ---------------------------------------------------------------------------
// Per-edge geometry: radial + coord_diff
// ---------------------------------------------------------------------------
__global__ void prep_edges_kernel(const float* __restrict__ x, const int* __restrict__ rowi,
                                  const int* __restrict__ coli, float* __restrict__ radial,
                                  float* __restrict__ cdiff) {
    int e = blockIdx.x * 256 + threadIdx.x;
    if (e >= N_EDGES) return;
    int r = rowi[e], c = coli[e];
    float dx = x[r * 3 + 0] - x[c * 3 + 0];
    float dy = x[r * 3 + 1] - x[c * 3 + 1];
    float dz = x[r * 3 + 2] - x[c * 3 + 2];
    float rad = dx * dx + dy * dy + dz * dz;
    radial[e] = rad;
    float inv = 1.0f / (sqrtf(rad + 1e-8f) + 1.0f);  // NORM_CONST = 1
    cdiff[e * 3 + 0] = dx * inv;
    cdiff[e * 3 + 1] = dy * inv;
    cdiff[e * 3 + 2] = dz * inv;
}

// ---------------------------------------------------------------------------
// Quantum precompute (parameter-only, tiny): Cayley transform via Gauss-Jordan,
// Kronecker gate, CNOT permutation, fold into one real 128x128 W_q per layer.
// ---------------------------------------------------------------------------
__global__ void herm_build_kernel(const float* __restrict__ Are, const float* __restrict__ Aim,
                                  cplx* __restrict__ aug, cplx* __restrict__ am) {
    int idx = blockIdx.x * 256 + threadIdx.x;
    if (idx >= 128 * 128) return;
    int r = idx >> 7, c = idx & 127;
    float hr = Are[r * 128 + c] + Are[c * 128 + r];
    float hi = Aim[r * 128 + c] - Aim[c * 128 + r];
    float d = (r == c) ? 1.0f : 0.0f;
    aug[r * 256 + c]       = { hr, hi + d };   // A + iI
    aug[r * 256 + 128 + c] = { d, 0.0f };      // identity
    am[idx]                = { hr, hi - d };   // A - iI
}

__global__ __launch_bounds__(256) void gauss_jordan_kernel(cplx* __restrict__ aug) {
    int tid = threadIdx.x;
    for (int k = 0; k < 128; ++k) {
        cplx piv = aug[k * 256 + k];
        float den = piv.re * piv.re + piv.im * piv.im;
        float ir = piv.re / den, ii = -piv.im / den;
        __syncthreads();
        {
            cplx v = aug[k * 256 + tid];
            aug[k * 256 + tid] = { v.re * ir - v.im * ii, v.re * ii + v.im * ir };
        }
        __syncthreads();
        if (tid < 128 && tid != k) {
            cplx f = aug[tid * 256 + k];
            for (int c = 0; c < 256; ++c) {
                cplx pk = aug[k * 256 + c];
                cplx v  = aug[tid * 256 + c];
                v.re -= f.re * pk.re - f.im * pk.im;
                v.im -= f.re * pk.im + f.im * pk.re;
                aug[tid * 256 + c] = v;
            }
        }
        __syncthreads();
    }
}

__global__ void cayley_mul_kernel(const cplx* __restrict__ am, const cplx* __restrict__ aug,
                                  cplx* __restrict__ Q) {
    int idx = blockIdx.x * 256 + threadIdx.x;
    if (idx >= 128 * 128) return;
    int r = idx >> 7, c = idx & 127;
    cplx s = { 0.0f, 0.0f };
    for (int k = 0; k < 128; ++k) {
        cplx a = am[r * 128 + k];
        cplx b = aug[k * 256 + 128 + c];
        s.re += a.re * b.re - a.im * b.im;
        s.im += a.re * b.im + a.im * b.re;
    }
    Q[idx] = s;
}

__global__ void build_perm_kernel(int* __restrict__ perm) {
    int i = threadIdx.x;
    if (i >= 128) return;
    const int cs[7] = {0, 1, 2, 3, 4, 5, 6};
    const int ts[7] = {1, 2, 3, 4, 5, 6, 0};
    int x = i;
    for (int p = 6; p >= 0; --p) {
        int cb = (x >> (6 - cs[p])) & 1;
        x ^= cb << (6 - ts[p]);
    }
    perm[i] = x;
}

__global__ void build_gk_kernel(const float* __restrict__ co, cplx* __restrict__ Gk) {
    int idx = blockIdx.x * 256 + threadIdx.x;
    if (idx >= 128 * 128) return;
    int r = idx >> 7, c = idx & 127;
    float a = co[0], b = co[1];
    float ca = cosf(a * 0.5f), sa = sinf(a * 0.5f);
    float cb = cosf(b * 0.5f), sb = sinf(b * 0.5f);
    cplx G00 = {  cb * ca,  sb * sa };
    cplx G01 = { -sb * ca, -cb * sa };
    cplx G10 = {  sb * ca, -cb * sa };
    cplx G11 = {  cb * ca, -sb * sa };
    cplx v = { 1.0f, 0.0f };
    for (int t = 0; t < 7; ++t) {
        int rb = (r >> (6 - t)) & 1, cbit = (c >> (6 - t)) & 1;
        cplx g = rb ? (cbit ? G11 : G10) : (cbit ? G01 : G00);
        cplx nv = { v.re * g.re - v.im * g.im, v.re * g.im + v.im * g.re };
        v = nv;
    }
    Gk[idx] = v;
}

__global__ void gkq_mul_kernel(const cplx* __restrict__ Gk, const int* __restrict__ perm,
                               const cplx* __restrict__ Q, cplx* __restrict__ Mp) {
    int idx = blockIdx.x * 256 + threadIdx.x;
    if (idx >= 128 * 128) return;
    int r = idx >> 7, c = idx & 127;
    int pr = perm[r];
    cplx s = { 0.0f, 0.0f };
    for (int k = 0; k < 128; ++k) {
        cplx a = Gk[pr * 128 + k];
        cplx b = Q[k * 128 + c];
        s.re += a.re * b.re - a.im * b.im;
        s.im += a.re * b.im + a.im * b.re;
    }
    Mp[idx] = s;
}

__global__ void cmatmul_kernel(const cplx* __restrict__ A, const cplx* __restrict__ B,
                               cplx* __restrict__ C) {
    int idx = blockIdx.x * 256 + threadIdx.x;
    if (idx >= 128 * 128) return;
    int r = idx >> 7, c = idx & 127;
    cplx s = { 0.0f, 0.0f };
    for (int k = 0; k < 128; ++k) {
        cplx a = A[r * 128 + k];
        cplx b = B[k * 128 + c];
        s.re += a.re * b.re - a.im * b.im;
        s.im += a.re * b.im + a.im * b.re;
    }
    C[idx] = s;
}

__global__ void make_wq_kernel(const cplx* __restrict__ C, const float* __restrict__ decw,
                               float* __restrict__ Wq) {
    int idx = blockIdx.x * 256 + threadIdx.x;
    if (idx >= 128 * 128) return;
    int k = idx >> 7, j = idx & 127;
    float s = 0.0f;
    for (int d = 0; d < 128; ++d) s += C[d * 128 + k].re * decw[d * 128 + j];
    Wq[idx] = s;
}

// ---------------------------------------------------------------------------
// Edge message MLP (one layer): 64 edges/block, 8 waves; each wave owns one
// 16-col slab and 4 row sub-tiles sharing every B-fragment (4x L2-traffic cut).
// h gathered as f16 via async global->LDS b128 copies (ASYNCcnt).
// ---------------------------------------------------------------------------
__global__ __launch_bounds__(256)
void edge_mlp_kernel(const _Float16* __restrict__ h16,
                     const int* __restrict__ rowi, const int* __restrict__ coli,
                     const float* __restrict__ radial, const float* __restrict__ eattr,
                     const _Float16* __restrict__ w1f, const float* __restrict__ w1raw,
                     const float* __restrict__ b1,
                     const _Float16* __restrict__ w2f, const float* __restrict__ b2,
                     float* __restrict__ agg) {
    __shared__ _Float16 Ain[64 * 264];   // 64 rows x 256 feats (+pad), f16
    __shared__ _Float16 Mid[64 * 136];
    __shared__ int   srcIdx[128];        // [0..63] row sources, [64..127] col sources
    __shared__ float ext0[64], ext1[64];

    const int tid = threadIdx.x;
    const int lane = tid & 31;
    const int w = tid >> 5;
    const int eb = blockIdx.x * 64;

    if (tid < 64) {
        int e = eb + tid;
        srcIdx[tid] = rowi[e];
        ext0[tid] = radial[e];
        ext1[tid] = eattr[e];
    } else if (tid < 128) {
        srcIdx[tid] = coli[eb + tid - 64];
    }
    if (eb + 64 < N_EDGES) {             // prefetch next block's gather rows
        if (tid < 64)       __builtin_prefetch(h16 + (size_t)rowi[eb + 64 + tid] * HF, 0, 3);
        else if (tid < 128) __builtin_prefetch(h16 + (size_t)coli[eb + tid] * HF, 0, 3);
    }
    __syncthreads();

    {   // async stage: 64 rows x 512B = 2048 x 16B chunks, 8 per thread
        uint32_t abase = (uint32_t)(uintptr_t)Ain;
#pragma unroll
        for (int i = 0; i < 8; ++i) {
            int c  = tid * 8 + i;
            int er = c >> 5;             // edge row 0..63
            int ci = c & 31;             // 16B chunk within 512B row
            int node = srcIdx[er + ((ci >> 4) << 6)];
            const void* g = h16 + (size_t)node * HF + (ci & 15) * 8;
            async_ld_b128(abase + er * 528 + ci * 16, g);
        }
        wait_asynccnt0();
    }
    __syncthreads();

    const int n0 = w * 16;
    const int col = lane & 15;
    const int rbase = (lane >> 4) << 3;

    v8f acc1[4] = {};
#pragma unroll
    for (int kt = 0; kt < 8; ++kt) {
        v16h b = load_b_frag(w1f, kt * 8 + w, lane);
#pragma unroll
        for (int s = 0; s < 4; ++s) {
            v16h a = load_a_frag(Ain + s * 16 * 264, 264, kt * 32, lane);
            acc1[s] = __builtin_amdgcn_wmma_f32_16x16x32_f16(false, a, false, b, (short)0, acc1[s], false, false);
        }
    }
    {
        float bb = b1[n0 + col];
        float wr = w1raw[256 * HF + n0 + col];
        float wa = w1raw[257 * HF + n0 + col];
#pragma unroll
        for (int s = 0; s < 4; ++s)
#pragma unroll
            for (int r = 0; r < 8; ++r) {
                int m = s * 16 + rbase + r;
                float v = acc1[s][r] + bb + ext0[m] * wr + ext1[m] * wa;
                Mid[m * 136 + n0 + col] = (_Float16)silu_f(v);
            }
    }
    __syncthreads();

    v8f acc2[4] = {};
#pragma unroll
    for (int kt = 0; kt < 4; ++kt) {
        v16h b = load_b_frag(w2f, kt * 8 + w, lane);
#pragma unroll
        for (int s = 0; s < 4; ++s) {
            v16h a = load_a_frag(Mid + s * 16 * 136, 136, kt * 32, lane);
            acc2[s] = __builtin_amdgcn_wmma_f32_16x16x32_f16(false, a, false, b, (short)0, acc2[s], false, false);
        }
    }
    {
        float bb = b2[n0 + col];
#pragma unroll
        for (int s = 0; s < 4; ++s)
#pragma unroll
            for (int r = 0; r < 8; ++r) {
                int m = s * 16 + rbase + r;
                float v = silu_f(acc2[s][r] + bb);
                atomicAdd(&agg[(size_t)srcIdx[m] * HF + n0 + col], v);
            }
    }
}

// ---------------------------------------------------------------------------
// Node update: q_in = [h, agg/100]@enc_w + enc_b ; normalize rows ;
// h_out = h_in + s @ W_q + dec_b. 16 nodes/block.
// ---------------------------------------------------------------------------
__global__ __launch_bounds__(256)
void node_update_kernel(const float* __restrict__ h_in, const float* __restrict__ agg,
                        const _Float16* __restrict__ encf, const float* __restrict__ encb,
                        const _Float16* __restrict__ wqf, const float* __restrict__ decb,
                        float* __restrict__ h_out) {
    __shared__ _Float16 Ain[16 * 264];
    __shared__ _Float16 Sm[16 * 136];
    __shared__ float norms[16];

    const int tid = threadIdx.x;
    const int lane = tid & 31;
    const int w = tid >> 5;
    const int nb = blockIdx.x * 16;

    {
        int r = tid >> 4;
        int f0 = (tid & 15) * 16;
        const float* src = (f0 < 128) ? (h_in + (size_t)(nb + r) * HF + f0)
                                      : (agg + (size_t)(nb + r) * HF + (f0 - 128));
        float sc = (f0 < 128) ? 1.0f : 0.01f;   // /NORM
        _Float16* dp = Ain + r * 264 + f0;
#pragma unroll
        for (int i = 0; i < 16; ++i) dp[i] = (_Float16)(src[i] * sc);
        if (tid < 16) norms[tid] = 0.0f;
    }
    __syncthreads();

    const int n0 = w * 16;
    const int col = lane & 15;
    const int rbase = (lane >> 4) << 3;

    v8f acc = {};
#pragma unroll
    for (int kt = 0; kt < 8; ++kt) {
        v16h a = load_a_frag(Ain, 264, kt * 32, lane);
        v16h b = load_b_frag(encf, kt * 8 + w, lane);
        acc = __builtin_amdgcn_wmma_f32_16x16x32_f16(false, a, false, b, (short)0, acc, false, false);
    }
    float qv[8];
    {
        float bb = encb[n0 + col];
#pragma unroll
        for (int r = 0; r < 8; ++r) {
            qv[r] = acc[r] + bb;
            atomicAdd(&norms[rbase + r], qv[r] * qv[r]);
        }
    }
    __syncthreads();
#pragma unroll
    for (int r = 0; r < 8; ++r) {
        int m = rbase + r;
        float inv = rsqrtf(norms[m] + 1e-12f);
        Sm[m * 136 + n0 + col] = (_Float16)(qv[r] * inv);
    }
    __syncthreads();

    v8f acc2 = {};
#pragma unroll
    for (int kt = 0; kt < 4; ++kt) {
        v16h a = load_a_frag(Sm, 136, kt * 32, lane);
        v16h b = load_b_frag(wqf, kt * 8 + w, lane);
        acc2 = __builtin_amdgcn_wmma_f32_16x16x32_f16(false, a, false, b, (short)0, acc2, false, false);
    }
    {
        float bb = decb[n0 + col];
#pragma unroll
        for (int r = 0; r < 8; ++r) {
            int m = rbase + r;
            size_t o = (size_t)(nb + m) * HF + n0 + col;
            h_out[o] = h_in[o] + acc2[r] + bb;
        }
    }
}

// ---------------------------------------------------------------------------
// Final coordinate MLP: 258->128->128->1; trans = coord_diff * t; atomic seg-sum.
// Same 64-edge/block structure as edge_mlp_kernel.
// ---------------------------------------------------------------------------
__global__ __launch_bounds__(256)
void coord_mlp_kernel(const _Float16* __restrict__ h16,
                      const int* __restrict__ rowi, const int* __restrict__ coli,
                      const float* __restrict__ radial, const float* __restrict__ eattr,
                      const _Float16* __restrict__ w1f, const float* __restrict__ w1raw,
                      const float* __restrict__ b1,
                      const _Float16* __restrict__ w2f, const float* __restrict__ b2,
                      const float* __restrict__ w3,
                      const float* __restrict__ cdiff, float* __restrict__ xacc) {
    __shared__ _Float16 Ain[64 * 264];
    __shared__ _Float16 Mid[64 * 136];
    __shared__ int   srcIdx[128];
    __shared__ float ext0[64], ext1[64], tacc[64];

    const int tid = threadIdx.x;
    const int lane = tid & 31;
    const int w = tid >> 5;
    const int eb = blockIdx.x * 64;

    if (tid < 64) {
        int e = eb + tid;
        srcIdx[tid] = rowi[e];
        ext0[tid] = radial[e];
        ext1[tid] = eattr[e];
        tacc[tid] = 0.0f;
    } else if (tid < 128) {
        srcIdx[tid] = coli[eb + tid - 64];
    }
    if (eb + 64 < N_EDGES) {
        if (tid < 64)       __builtin_prefetch(h16 + (size_t)rowi[eb + 64 + tid] * HF, 0, 3);
        else if (tid < 128) __builtin_prefetch(h16 + (size_t)coli[eb + tid] * HF, 0, 3);
    }
    __syncthreads();

    {
        uint32_t abase = (uint32_t)(uintptr_t)Ain;
#pragma unroll
        for (int i = 0; i < 8; ++i) {
            int c  = tid * 8 + i;
            int er = c >> 5;
            int ci = c & 31;
            int node = srcIdx[er + ((ci >> 4) << 6)];
            const void* g = h16 + (size_t)node * HF + (ci & 15) * 8;
            async_ld_b128(abase + er * 528 + ci * 16, g);
        }
        wait_asynccnt0();
    }
    __syncthreads();

    const int n0 = w * 16;
    const int col = lane & 15;
    const int rbase = (lane >> 4) << 3;

    v8f acc1[4] = {};
#pragma unroll
    for (int kt = 0; kt < 8; ++kt) {
        v16h b = load_b_frag(w1f, kt * 8 + w, lane);
#pragma unroll
        for (int s = 0; s < 4; ++s) {
            v16h a = load_a_frag(Ain + s * 16 * 264, 264, kt * 32, lane);
            acc1[s] = __builtin_amdgcn_wmma_f32_16x16x32_f16(false, a, false, b, (short)0, acc1[s], false, false);
        }
    }
    {
        float bb = b1[n0 + col];
        float wr = w1raw[256 * HF + n0 + col];
        float wa = w1raw[257 * HF + n0 + col];
#pragma unroll
        for (int s = 0; s < 4; ++s)
#pragma unroll
            for (int r = 0; r < 8; ++r) {
                int m = s * 16 + rbase + r;
                float v = acc1[s][r] + bb + ext0[m] * wr + ext1[m] * wa;
                Mid[m * 136 + n0 + col] = (_Float16)silu_f(v);
            }
    }
    __syncthreads();

    v8f acc2[4] = {};
#pragma unroll
    for (int kt = 0; kt < 4; ++kt) {
        v16h b = load_b_frag(w2f, kt * 8 + w, lane);
#pragma unroll
        for (int s = 0; s < 4; ++s) {
            v16h a = load_a_frag(Mid + s * 16 * 136, 136, kt * 32, lane);
            acc2[s] = __builtin_amdgcn_wmma_f32_16x16x32_f16(false, a, false, b, (short)0, acc2[s], false, false);
        }
    }
    {
        float bb = b2[n0 + col];
        float w3v = w3[n0 + col];
#pragma unroll
        for (int s = 0; s < 4; ++s)
#pragma unroll
            for (int r = 0; r < 8; ++r) {
                float v = silu_f(acc2[s][r] + bb) * w3v;
                atomicAdd(&tacc[s * 16 + rbase + r], v);
            }
    }
    __syncthreads();
    if (tid < 64) {
        int e = eb + tid;
        float t = tacc[tid];
        int rr = srcIdx[tid];
        atomicAdd(&xacc[rr * 3 + 0], cdiff[e * 3 + 0] * t);
        atomicAdd(&xacc[rr * 3 + 1], cdiff[e * 3 + 1] * t);
        atomicAdd(&xacc[rr * 3 + 2], cdiff[e * 3 + 2] * t);
    }
}

__global__ void finalize_x_kernel(const float* __restrict__ x, const float* __restrict__ xacc,
                                  float* __restrict__ xout) {
    int i = blockIdx.x * 256 + threadIdx.x;
    if (i < N_NODES * 3) xout[i] = x[i] + xacc[i] * 0.01f;   // /NORM
}

// ---------------------------------------------------------------------------
extern "C" void kernel_launch(void* const* d_in, const int* in_sizes, int n_in,
                              void* d_out, int out_size, void* d_ws, size_t ws_size,
                              hipStream_t stream) {
    const float* h0     = (const float*)d_in[0];
    const float* x      = (const float*)d_in[1];
    const int*   ei     = (const int*)d_in[2];
    const float* eattr  = (const float*)d_in[3];
    const float* e_w1   = (const float*)d_in[4];
    const float* e_b1   = (const float*)d_in[5];
    const float* e_w2   = (const float*)d_in[6];
    const float* e_b2   = (const float*)d_in[7];
    const float* enc_w  = (const float*)d_in[8];
    const float* enc_b  = (const float*)d_in[9];
    const float* coeffs = (const float*)d_in[10];
    const float* A_re   = (const float*)d_in[11];
    const float* A_im   = (const float*)d_in[12];
    const float* dec_w  = (const float*)d_in[13];
    const float* dec_b  = (const float*)d_in[14];
    const float* c_w1   = (const float*)d_in[15];
    const float* c_b1   = (const float*)d_in[16];
    const float* c_w2   = (const float*)d_in[17];
    const float* c_b2   = (const float*)d_in[18];
    const float* c_w3   = (const float*)d_in[19];

    const int* rowi = ei;
    const int* coli = ei + N_EDGES;

    char* ws = (char*)d_ws;
    size_t off = 0;
    auto alloc = [&](size_t bytes) -> char* {
        char* p = ws + off;
        off += (bytes + 255) & ~(size_t)255;
        return p;
    };

    float*    radial = (float*)alloc((size_t)N_EDGES * 4);
    float*    cdiff  = (float*)alloc((size_t)N_EDGES * 12);
    float*    agg    = (float*)alloc((size_t)N_NODES * HF * 4);
    float*    h_cur  = (float*)alloc((size_t)N_NODES * HF * 4);
    float*    xacc   = (float*)alloc((size_t)N_NODES * 12);
    _Float16* h16    = (_Float16*)alloc((size_t)N_NODES * HF * 2);

    _Float16* ew1f[2] = { (_Float16*)alloc(32768 * 2), (_Float16*)alloc(32768 * 2) };
    _Float16* ew2f[2] = { (_Float16*)alloc(16384 * 2), (_Float16*)alloc(16384 * 2) };
    _Float16* encf[2] = { (_Float16*)alloc(32768 * 2), (_Float16*)alloc(32768 * 2) };
    _Float16* wqf[2]  = { (_Float16*)alloc(16384 * 2), (_Float16*)alloc(16384 * 2) };
    _Float16* cw1f    = (_Float16*)alloc(32768 * 2);
    _Float16* cw2f    = (_Float16*)alloc(16384 * 2);

    cplx* aug = (cplx*)alloc(128 * 256 * sizeof(cplx));
    cplx* am  = (cplx*)alloc(128 * 128 * sizeof(cplx));
    cplx* Qc  = (cplx*)alloc(128 * 128 * sizeof(cplx));
    cplx* Mp0 = (cplx*)alloc(128 * 128 * sizeof(cplx));
    cplx* Mp1 = (cplx*)alloc(128 * 128 * sizeof(cplx));
    cplx* Cc  = (cplx*)alloc(128 * 128 * sizeof(cplx));
    cplx* Gk  = (cplx*)alloc(128 * 128 * sizeof(cplx));
    int*  perm = (int*)alloc(128 * sizeof(int));
    float* Wq[2] = { (float*)alloc(128 * 128 * 4), (float*)alloc(128 * 128 * 4) };

    // --- geometry + weight swizzles ---
    prep_edges_kernel<<<(N_EDGES + 255) / 256, 256, 0, stream>>>(x, rowi, coli, radial, cdiff);
    for (int l = 0; l < 2; ++l) {
        make_bfrag_kernel<<<(8 * 8 * 512 + 255) / 256, 256, 0, stream>>>(e_w1 + (size_t)l * 258 * HF, ew1f[l], 8, 8, HF);
        make_bfrag_kernel<<<(4 * 8 * 512 + 255) / 256, 256, 0, stream>>>(e_w2 + (size_t)l * HF * HF, ew2f[l], 4, 8, HF);
        make_bfrag_kernel<<<(8 * 8 * 512 + 255) / 256, 256, 0, stream>>>(enc_w + (size_t)l * 256 * HF, encf[l], 8, 8, HF);
    }
    make_bfrag_kernel<<<(8 * 8 * 512 + 255) / 256, 256, 0, stream>>>(c_w1, cw1f, 8, 8, HF);
    make_bfrag_kernel<<<(4 * 8 * 512 + 255) / 256, 256, 0, stream>>>(c_w2, cw2f, 4, 8, HF);

    // --- quantum parameter precompute: W_q[l] = Re(Mp1 @ Mp0)^T @ dec_w[l] ---
    build_perm_kernel<<<1, 128, 0, stream>>>(perm);
    for (int l = 0; l < 2; ++l) {
        for (int j = 0; j < 2; ++j) {
            int lj = l * 2 + j;
            herm_build_kernel<<<64, 256, 0, stream>>>(A_re + (size_t)lj * 16384, A_im + (size_t)lj * 16384, aug, am);
            gauss_jordan_kernel<<<1, 256, 0, stream>>>(aug);
            cayley_mul_kernel<<<64, 256, 0, stream>>>(am, aug, Qc);
            build_gk_kernel<<<64, 256, 0, stream>>>(coeffs + (size_t)lj * 2, Gk);
            gkq_mul_kernel<<<64, 256, 0, stream>>>(Gk, perm, Qc, (j == 0) ? Mp0 : Mp1);
        }
        cmatmul_kernel<<<64, 256, 0, stream>>>(Mp1, Mp0, Cc);
        make_wq_kernel<<<64, 256, 0, stream>>>(Cc, dec_w + (size_t)l * 16384, Wq[l]);
        make_bfrag_kernel<<<(4 * 8 * 512 + 255) / 256, 256, 0, stream>>>(Wq[l], wqf[l], 4, 8, HF);
    }

    // --- main GNN layers ---
    float* hout = (float*)d_out;                          // [N,128]
    float* xout = (float*)d_out + (size_t)N_NODES * HF;   // [N,3]

    const float* hin = h0;
    for (int l = 0; l < 2; ++l) {
        zero_kernel<<<(N_NODES * HF + 255) / 256, 256, 0, stream>>>(agg, N_NODES * HF);
        to_half_kernel<<<(N_NODES * HF + 255) / 256, 256, 0, stream>>>(hin, h16, N_NODES * HF);
        edge_mlp_kernel<<<N_EDGES / 64, 256, 0, stream>>>(
            h16, rowi, coli, radial, eattr,
            ew1f[l], e_w1 + (size_t)l * 258 * HF, e_b1 + (size_t)l * HF,
            ew2f[l], e_b2 + (size_t)l * HF, agg);
        float* hnext = (l == 0) ? h_cur : hout;
        node_update_kernel<<<N_NODES / 16, 256, 0, stream>>>(
            hin, agg, encf[l], enc_b + (size_t)l * HF, wqf[l], dec_b + (size_t)l * HF, hnext);
        hin = hnext;
    }

    // --- coordinate head ---
    zero_kernel<<<(N_NODES * 3 + 255) / 256, 256, 0, stream>>>(xacc, N_NODES * 3);
    to_half_kernel<<<(N_NODES * HF + 255) / 256, 256, 0, stream>>>(hout, h16, N_NODES * HF);
    coord_mlp_kernel<<<N_EDGES / 64, 256, 0, stream>>>(
        h16, rowi, coli, radial, eattr,
        cw1f, c_w1, c_b1, cw2f, c_b2, c_w3, cdiff, xacc);
    finalize_x_kernel<<<(N_NODES * 3 + 255) / 256, 256, 0, stream>>>(x, xacc, xout);
}
